// CrossAttention_38019050504962
// MI455X (gfx1250) — matrix operand order repaired and checked by reading, not compiled, not run on hardware
//
#include <hip/hip_runtime.h>

typedef __attribute__((ext_vector_type(16))) _Float16 v16h;
typedef __attribute__((ext_vector_type(8)))  _Float16 v8h;
typedef __attribute__((ext_vector_type(4)))  _Float16 v4h;
typedef __attribute__((ext_vector_type(8)))  float    v8f;

#define N_DIM  4
#define B_DIM  32
#define F_DIM  1024
#define H_DIM  4
#define DH_DIM 256
#define R_DIM  128   // N*B
#define FF_DIM 4096  // 4*F

// ---------------- f32 -> f16 weight conversion (4 elems/thread) ---------------------
__global__ void cvt_f16_kernel(const float* __restrict__ src, _Float16* __restrict__ dst) {
    const size_t i = ((size_t)blockIdx.x * 256 + threadIdx.x) * 4;
    const float4 s = *reinterpret_cast<const float4*>(src + i);
    v4h d;
    d[0] = (_Float16)s.x; d[1] = (_Float16)s.y; d[2] = (_Float16)s.z; d[3] = (_Float16)s.w;
    *reinterpret_cast<v4h*>(dst + i) = d;
}

// ---------------- LayerNorm: one block (256 threads) per row of 1024 ----------------
__global__ void ln_kernel(const float* __restrict__ x, const float* __restrict__ g,
                          const float* __restrict__ b, float* __restrict__ out) {
    __shared__ float s_sum[256], s_sq[256];
    __shared__ float s_mean, s_rstd;
    const int row = blockIdx.x;
    const int tid = threadIdx.x;
    const float* xr = x + (size_t)row * F_DIM;
    float vals[4];
    float lsum = 0.f, lsq = 0.f;
#pragma unroll
    for (int i = 0; i < 4; ++i) {
        float v = xr[tid + i * 256];
        vals[i] = v; lsum += v; lsq += v * v;
    }
    s_sum[tid] = lsum; s_sq[tid] = lsq;
    __syncthreads();
    for (int s = 128; s > 0; s >>= 1) {
        if (tid < s) { s_sum[tid] += s_sum[tid + s]; s_sq[tid] += s_sq[tid + s]; }
        __syncthreads();
    }
    if (tid == 0) {
        float m   = s_sum[0] * (1.f / F_DIM);
        float var = s_sq[0] * (1.f / F_DIM) - m * m;
        s_mean = m;
        s_rstd = rsqrtf(var + 1e-5f);
    }
    __syncthreads();
    const float m = s_mean, rs = s_rstd;
    float* outr = out + (size_t)row * F_DIM;
#pragma unroll
    for (int i = 0; i < 4; ++i) {
        int c = tid + i * 256;
        outr[c] = (vals[i] - m) * rs * g[c] + b[c];
    }
}

// -------- pack helpers --------------------------------------------------------------
__device__ __forceinline__ v16h pack16(float4 a, float4 b, float4 c, float4 d) {
    v16h r;
    r[0]=(_Float16)a.x;  r[1]=(_Float16)a.y;  r[2]=(_Float16)a.z;  r[3]=(_Float16)a.w;
    r[4]=(_Float16)b.x;  r[5]=(_Float16)b.y;  r[6]=(_Float16)b.z;  r[7]=(_Float16)b.w;
    r[8]=(_Float16)c.x;  r[9]=(_Float16)c.y;  r[10]=(_Float16)c.z; r[11]=(_Float16)c.w;
    r[12]=(_Float16)d.x; r[13]=(_Float16)d.y; r[14]=(_Float16)d.z; r[15]=(_Float16)d.w;
    return r;
}
__device__ __forceinline__ v16h cat8(v8h lo, v8h hi) {
    return __builtin_shufflevector(lo, hi, 0,1,2,3,4,5,6,7,8,9,10,11,12,13,14,15);
}

// Raw per-K-step operand data (one 16x32 A tile + four 16x32 B tiles, f16 B)
struct KStep {
    float4 a0, a1, a2, a3;
    v8h b0l, b0h, b1l, b1h, b2l, b2h, b3l, b3h;
};

// ---------------- WMMA GEMM: C[R,O] = A[R,K] @ Wh[O,K]^T + bias (+relu) (+addend) ---
// A: f32 activations (converted in-register), Wh: pre-converted f16 weights.
// 4 waves/block; each wave computes a 16x64 tile; explicit register double-buffering:
// next K-step's loads are issued before the current step's 4 WMMAs.
template <bool RELU, bool ADD>
__global__ void gemm_wmma(const float* __restrict__ A, const _Float16* __restrict__ Wh,
                          const float* __restrict__ bias, const float* __restrict__ addend,
                          float* __restrict__ C, int K, int O) {
    const int lane = threadIdx.x & 31;
    const int warp = threadIdx.x >> 5;       // 0..3
    const int hf   = lane >> 4;              // wave half: 0 or 1
    const int l16  = lane & 15;
    const int rowBase = blockIdx.y * 16;
    const int colBase = blockIdx.x * 256 + warp * 64;

    // A-frag: row = rowBase + l16; 16 contiguous K at +16*hf
    const float* pa = A + (size_t)(rowBase + l16) * K + 16 * hf;
    // B-frags: weight rows o = colBase + 16*t + l16; 16 f16 at +16*hf
    const _Float16* pw0 = Wh + (size_t)(colBase +  0 + l16) * K + 16 * hf;
    const _Float16* pw1 = Wh + (size_t)(colBase + 16 + l16) * K + 16 * hf;
    const _Float16* pw2 = Wh + (size_t)(colBase + 32 + l16) * K + 16 * hf;
    const _Float16* pw3 = Wh + (size_t)(colBase + 48 + l16) * K + 16 * hf;

    auto load_step = [&](int kk) {
        KStep s;
        const float4* pa4 = reinterpret_cast<const float4*>(pa + kk);
        s.a0 = pa4[0]; s.a1 = pa4[1]; s.a2 = pa4[2]; s.a3 = pa4[3];
        const v8h* r0 = reinterpret_cast<const v8h*>(pw0 + kk);
        const v8h* r1 = reinterpret_cast<const v8h*>(pw1 + kk);
        const v8h* r2 = reinterpret_cast<const v8h*>(pw2 + kk);
        const v8h* r3 = reinterpret_cast<const v8h*>(pw3 + kk);
        s.b0l = r0[0]; s.b0h = r0[1];
        s.b1l = r1[0]; s.b1h = r1[1];
        s.b2l = r2[0]; s.b2h = r2[1];
        s.b3l = r3[0]; s.b3h = r3[1];
        // stream-ahead prefetch of the weight operand (global_prefetch_b8)
        __builtin_prefetch(pw0 + kk + 96, 0, 1);
        __builtin_prefetch(pw1 + kk + 96, 0, 1);
        __builtin_prefetch(pw2 + kk + 96, 0, 1);
        __builtin_prefetch(pw3 + kk + 96, 0, 1);
        return s;
    };

    v8f acc0 = {}, acc1 = {}, acc2 = {}, acc3 = {};
    auto compute = [&](const KStep& s) {
        const v16h af  = pack16(s.a0, s.a1, s.a2, s.a3);
        const v16h bf0 = cat8(s.b0l, s.b0h);
        const v16h bf1 = cat8(s.b1l, s.b1h);
        const v16h bf2 = cat8(s.b2l, s.b2h);
        const v16h bf3 = cat8(s.b3l, s.b3h);
        acc0 = __builtin_amdgcn_wmma_f32_16x16x32_f16(false, af, false, bf0, (short)0, acc0, false, false);
        acc1 = __builtin_amdgcn_wmma_f32_16x16x32_f16(false, af, false, bf1, (short)0, acc1, false, false);
        acc2 = __builtin_amdgcn_wmma_f32_16x16x32_f16(false, af, false, bf2, (short)0, acc2, false, false);
        acc3 = __builtin_amdgcn_wmma_f32_16x16x32_f16(false, af, false, bf3, (short)0, acc3, false, false);
    };

    // software pipeline: loads for step kk+32 are in flight during WMMAs of step kk
    KStep cur = load_step(0);
#pragma unroll 2
    for (int kk = 0; kk < K - 32; kk += 32) {
        KStep nxt = load_step(kk + 32);
        compute(cur);
        cur = nxt;
    }
    compute(cur);

    const v8f accs[4] = {acc0, acc1, acc2, acc3};
#pragma unroll
    for (int t = 0; t < 4; ++t) {
        const int col = colBase + t * 16 + l16;
        const float bval = bias[col];
#pragma unroll
        for (int v = 0; v < 8; ++v) {
            const int row = rowBase + v + 8 * hf;
            const size_t idx = (size_t)row * O + col;
            float val = accs[t][v] + bval;
            if constexpr (RELU) val = fmaxf(val, 0.f);
            if constexpr (ADD)  val += addend[idx];
            C[idx] = val;
        }
    }
}

// ---------------- Factorized outer-product attention -------------------------------
// att[i,b,h,d] = sum_{j!=i} ( sum_e exp(t*k_e)*v_e ) / ( sum_e exp(t*k_e) ),  t = q_d/16
__global__ void attn_kernel(const float* __restrict__ q, const float* __restrict__ k,
                            const float* __restrict__ v, float* __restrict__ att) {
    __shared__ float sk[DH_DIM], sv[DH_DIM], red[256];
    __shared__ float s_kmax, s_kmin;
    const int gid = blockIdx.x;
    const int i = gid / (B_DIM * H_DIM);
    const int rem = gid % (B_DIM * H_DIM);
    const int b = rem / H_DIM;
    const int h = rem % H_DIM;
    const int d = threadIdx.x;

    const float qv = q[(size_t)(i * B_DIM + b) * F_DIM + h * DH_DIM + d];
    const float t = qv * 0.0625f; // 1/sqrt(DH), DH=256
    float accv = 0.f;

    for (int j = 0; j < N_DIM; ++j) {
        if (j == i) continue;
        const size_t base = (size_t)(j * B_DIM + b) * F_DIM + h * DH_DIM;
        const float kv = k[base + d];
        sk[d] = kv;
        sv[d] = v[base + d];
        // block max of k
        red[d] = kv; __syncthreads();
        for (int s = 128; s > 0; s >>= 1) {
            if (d < s) red[d] = fmaxf(red[d], red[d + s]);
            __syncthreads();
        }
        if (d == 0) s_kmax = red[0];
        __syncthreads();
        // block min of k
        red[d] = kv; __syncthreads();
        for (int s = 128; s > 0; s >>= 1) {
            if (d < s) red[d] = fminf(red[d], red[d + s]);
            __syncthreads();
        }
        if (d == 0) s_kmin = red[0];
        __syncthreads();

        const float m = (t >= 0.f) ? t * s_kmax : t * s_kmin; // = max_e t*k_e
        float den = 0.f, num = 0.f;
        for (int e = 0; e < DH_DIM; ++e) {
            float x = __expf(t * sk[e] - m);
            den += x;
            num += x * sv[e];
        }
        accv += num / den;
        __syncthreads(); // before sk/sv reuse
    }
    att[(size_t)(i * B_DIM + b) * F_DIM + h * DH_DIM + d] = accv;
}

extern "C" void kernel_launch(void* const* d_in, const int* in_sizes, int n_in,
                              void* d_out, int out_size, void* d_ws, size_t ws_size,
                              hipStream_t stream) {
    const float* features = (const float*)d_in[0];
    const float* Wq  = (const float*)d_in[1];
    const float* bq  = (const float*)d_in[2];
    const float* Wk  = (const float*)d_in[3];
    const float* bk  = (const float*)d_in[4];
    const float* Wv  = (const float*)d_in[5];
    const float* bv  = (const float*)d_in[6];
    const float* Wo  = (const float*)d_in[7];
    const float* bo  = (const float*)d_in[8];
    const float* g1  = (const float*)d_in[9];
    const float* b1  = (const float*)d_in[10];
    const float* g2  = (const float*)d_in[11];
    const float* b2  = (const float*)d_in[12];
    const float* W1  = (const float*)d_in[13];
    const float* bf1 = (const float*)d_in[14];
    const float* W2  = (const float*)d_in[15];
    const float* bf2 = (const float*)d_in[16];
    float* out = (float*)d_out;

    // ---- workspace layout: f32 activations, then f16 weight copies ----
    float* ws = (float*)d_ws;
    float* xq       = ws;                       // 128*1024
    float* q        = xq       + R_DIM * F_DIM;
    float* kbuf     = q        + R_DIM * F_DIM;
    float* vbuf     = kbuf     + R_DIM * F_DIM;
    float* att      = vbuf     + R_DIM * F_DIM;
    float* attn_out = att      + R_DIM * F_DIM;
    float* y        = attn_out + R_DIM * F_DIM;
    float* h1       = y        + R_DIM * F_DIM; // 128*4096
    _Float16* wh = (_Float16*)(h1 + (size_t)R_DIM * FF_DIM);
    _Float16* Whq = wh;                                  // F*F
    _Float16* Whk = Whq + (size_t)F_DIM * F_DIM;
    _Float16* Whv = Whk + (size_t)F_DIM * F_DIM;
    _Float16* Who = Whv + (size_t)F_DIM * F_DIM;
    _Float16* Wh1 = Who + (size_t)F_DIM * F_DIM;         // FF*F
    _Float16* Wh2 = Wh1 + (size_t)FF_DIM * F_DIM;        // F*FF

    const int nFF  = F_DIM * F_DIM;     // 1M
    const int nFFN = FF_DIM * F_DIM;    // 4M

    // ---- weight conversion passes (f32 -> f16, each elem once) ----
    cvt_f16_kernel<<<nFF  / 1024, 256, 0, stream>>>(Wq, Whq);
    cvt_f16_kernel<<<nFF  / 1024, 256, 0, stream>>>(Wk, Whk);
    cvt_f16_kernel<<<nFF  / 1024, 256, 0, stream>>>(Wv, Whv);
    cvt_f16_kernel<<<nFF  / 1024, 256, 0, stream>>>(Wo, Who);
    cvt_f16_kernel<<<nFFN / 1024, 256, 0, stream>>>(W1, Wh1);
    cvt_f16_kernel<<<nFFN / 1024, 256, 0, stream>>>(W2, Wh2);

    const dim3 blk(128, 1, 1);
    const dim3 gF(F_DIM / 256, R_DIM / 16);     // O = 1024
    const dim3 gFF(FF_DIM / 256, R_DIM / 16);   // O = 4096

    // norm1 (queries only)
    ln_kernel<<<R_DIM, 256, 0, stream>>>(features, g1, b1, xq);
    // projections
    gemm_wmma<false, false><<<gF, blk, 0, stream>>>(xq,       Whq, bq, nullptr, q,    F_DIM, F_DIM);
    gemm_wmma<false, false><<<gF, blk, 0, stream>>>(features, Whk, bk, nullptr, kbuf, F_DIM, F_DIM);
    gemm_wmma<false, false><<<gF, blk, 0, stream>>>(features, Whv, bv, nullptr, vbuf, F_DIM, F_DIM);
    // factorized outer-product attention with j!=i mask
    attn_kernel<<<N_DIM * B_DIM * H_DIM, 256, 0, stream>>>(q, kbuf, vbuf, att);
    // output projection + residual (xq)
    gemm_wmma<false, true><<<gF, blk, 0, stream>>>(att, Who, bo, xq, attn_out, F_DIM, F_DIM);
    // norm2
    ln_kernel<<<R_DIM, 256, 0, stream>>>(attn_out, g2, b2, y);
    // FFN
    gemm_wmma<true,  false><<<gFF, blk, 0, stream>>>(y,  Wh1, bf1, nullptr,  h1,  F_DIM,  FF_DIM);
    gemm_wmma<false, true ><<<gF,  blk, 0, stream>>>(h1, Wh2, bf2, attn_out, out, FF_DIM, F_DIM);
}
